// HierarchicalGraphTransformer_32392643346794
// MI455X (gfx1250) — compile-verified, matrix operand order checked by reading, not run on hardware
//
#include <hip/hip_runtime.h>
#include <hip/hip_bf16.h>

// ---------------------------------------------------------------------------
// HierarchicalGraphTransformer for MI455X (gfx1250), wave32 + WMMA bf16 + TDM.
// N=4096, E=262144, H=4, C=256, D=256, FF=2048, L=2
// ---------------------------------------------------------------------------

typedef __attribute__((ext_vector_type(16))) __bf16   v16bf;
typedef __attribute__((ext_vector_type(8)))  float    v8f;
typedef __attribute__((ext_vector_type(4)))  unsigned u32x4;
typedef __attribute__((ext_vector_type(8)))  int      i32x8;
typedef __attribute__((ext_vector_type(4)))  int      i32x4;

#define GN   4096
#define GE   262144
#define GH   4
#define GC   256
#define GD   256
#define GFF  2048
#define GL   2
#define DH   64

// ------------------------- WMMA fragment loaders ---------------------------
// 16-bit A-operand layout (16x32, MxK), wave32:
//   lanes 0-15 : M = lane,     halves 0..7 = K 0..7,  halves 8..15 = K 16..23
//   lanes 16-31: M = lane-16,  halves 0..7 = K 8..15, halves 8..15 = K 24..31
__device__ __forceinline__ v16bf frag_rowK(const float* rowPtr, int kBase, int lane) {
  int koff = (lane < 16) ? 0 : 8;
  v16bf f;
#pragma unroll
  for (int i = 0; i < 8; ++i) f[i]     = (__bf16)rowPtr[kBase + koff + i];
#pragma unroll
  for (int i = 0; i < 8; ++i) f[i + 8] = (__bf16)rowPtr[kBase + 16 + koff + i];
  return f;
}

// B-operand when B is stored [K,Nc] row-major: column fixed per lane, K strided.
__device__ __forceinline__ v16bf frag_colK(const float* colPtr, int ld, int kBase, int lane) {
  int koff = (lane < 16) ? 0 : 8;
  v16bf f;
#pragma unroll
  for (int i = 0; i < 8; ++i) f[i]     = (__bf16)colPtr[(size_t)(kBase + koff + i) * ld];
#pragma unroll
  for (int i = 0; i < 8; ++i) f[i + 8] = (__bf16)colPtr[(size_t)(kBase + 16 + koff + i) * ld];
  return f;
}

// ------------------------------ GEMM (WMMA) --------------------------------
// Cout[M,Nc] = act( A[M,K] @ B(^T) + bias (+ residual) )
// btrans=1: B stored [Nc,K] row-major  (x @ W^T)
// btrans=0: B stored [K,Nc] row-major  (x @ W)
// Workgroup: 4 waves; tile M=16 x N=256 (wave -> 4 sub-tiles of 16).
// A[16,K] panel staged once into LDS by the Tensor Data Mover (TDM);
// each wave reuses it for 4 WMMA per K-step.
__global__ void gemm_bf16_wmma(const float* __restrict__ A, const float* __restrict__ B,
                               const float* __restrict__ bias, const float* __restrict__ residual,
                               float* __restrict__ Cout,
                               int M, int Nc, int K, int btrans, int relu)
{
  extern __shared__ float Ash[];                  // 16 * K floats
  const int lane   = threadIdx.x & 31;
  const int wave   = threadIdx.x >> 5;
  const int mBase  = blockIdx.x * 16;
  const int nBlock = blockIdx.y * 256;

  // ---- stage A[16,K] row-panel into LDS via TENSOR_LOAD_TO_LDS ----
#if __has_builtin(__builtin_amdgcn_tensor_load_to_lds) && __has_builtin(__builtin_amdgcn_s_wait_tensorcnt)
  if (wave == 0) {
    const unsigned long long ga =
        (unsigned long long)(const void*)(A + (size_t)mBase * K);
    const unsigned ldsOff = (unsigned)(unsigned long long)(void*)Ash; // low 32b of flat = LDS addr
    // D# group 0: count=1 | lds_addr | global_addr[56:0] | type=2
    u32x4 g0 = { 1u,
                 ldsOff,
                 (unsigned)(ga & 0xFFFFFFFFull),
                 (unsigned)((ga >> 32) & 0x01FFFFFFull) | (2u << 30) };
    // D# group 1: data_size=4B; tensor_dim0=K, tensor_dim1=16;
    //             tile_dim0=K, tile_dim1=16; tensor_dim0_stride=K
    i32x8 g1 = { (int)(2u << 16),
                 (int)(((unsigned)K & 0xFFFFu) << 16),
                 (int)((((unsigned)K >> 16) & 0xFFFFu) | (16u << 16)),
                 (int)(((unsigned)K & 0xFFFFu) << 16),
                 16,
                 K,
                 0, 0 };
    i32x4 gz4 = { 0, 0, 0, 0 };
    i32x8 gz8 = { 0, 0, 0, 0, 0, 0, 0, 0 };
    __builtin_amdgcn_tensor_load_to_lds(g0, g1, gz4, gz4, gz8, 0);
    __builtin_amdgcn_s_wait_tensorcnt(0);
  }
#else
  for (int j = threadIdx.x; j < 16 * K; j += 128)
    Ash[j] = A[(size_t)(mBase + j / K) * K + (j % K)];
#endif
  __syncthreads();

  const float* rowAsh = Ash + (size_t)(lane & 15) * K;

  v8f acc0 = {}, acc1 = {}, acc2 = {}, acc3 = {};
  const int n0 = nBlock + (wave * 4) * 16;

  if (btrans) {
    const float* w0 = B + (size_t)(n0 +  0 + (lane & 15)) * K;
    const float* w1 = B + (size_t)(n0 + 16 + (lane & 15)) * K;
    const float* w2 = B + (size_t)(n0 + 32 + (lane & 15)) * K;
    const float* w3 = B + (size_t)(n0 + 48 + (lane & 15)) * K;
    for (int k = 0; k < K; k += 32) {
      __builtin_prefetch(w0 + k + 128, 0, 1);     // global_prefetch_b8
      v16bf a = frag_rowK(rowAsh, k, lane);       // ds_load path
      acc0 = __builtin_amdgcn_wmma_f32_16x16x32_bf16(false, a, false, frag_rowK(w0, k, lane), (short)0, acc0, false, false);
      acc1 = __builtin_amdgcn_wmma_f32_16x16x32_bf16(false, a, false, frag_rowK(w1, k, lane), (short)0, acc1, false, false);
      acc2 = __builtin_amdgcn_wmma_f32_16x16x32_bf16(false, a, false, frag_rowK(w2, k, lane), (short)0, acc2, false, false);
      acc3 = __builtin_amdgcn_wmma_f32_16x16x32_bf16(false, a, false, frag_rowK(w3, k, lane), (short)0, acc3, false, false);
    }
  } else {
    const float* w0 = B + (n0 +  0 + (lane & 15));
    const float* w1 = B + (n0 + 16 + (lane & 15));
    const float* w2 = B + (n0 + 32 + (lane & 15));
    const float* w3 = B + (n0 + 48 + (lane & 15));
    for (int k = 0; k < K; k += 32) {
      v16bf a = frag_rowK(rowAsh, k, lane);
      acc0 = __builtin_amdgcn_wmma_f32_16x16x32_bf16(false, a, false, frag_colK(w0, Nc, k, lane), (short)0, acc0, false, false);
      acc1 = __builtin_amdgcn_wmma_f32_16x16x32_bf16(false, a, false, frag_colK(w1, Nc, k, lane), (short)0, acc1, false, false);
      acc2 = __builtin_amdgcn_wmma_f32_16x16x32_bf16(false, a, false, frag_colK(w2, Nc, k, lane), (short)0, acc2, false, false);
      acc3 = __builtin_amdgcn_wmma_f32_16x16x32_bf16(false, a, false, frag_colK(w3, Nc, k, lane), (short)0, acc3, false, false);
    }
  }

  const int rofs = (lane < 16) ? 0 : 8;
  v8f accs[4] = { acc0, acc1, acc2, acc3 };
#pragma unroll
  for (int j = 0; j < 4; ++j) {
    const int col = n0 + j * 16 + (lane & 15);
    const float bv = bias ? bias[col] : 0.f;
#pragma unroll
    for (int r = 0; r < 8; ++r) {
      int row = mBase + rofs + r;
      float v = accs[j][r] + bv;
      if (residual) v += residual[(size_t)row * Nc + col];
      if (relu && v < 0.f) v = 0.f;
      Cout[(size_t)row * Nc + col] = v;
    }
  }
}

// ------------------------- fused attention (WMMA) --------------------------
// One workgroup (4 waves) per (16 query rows, head).  Full 16x4096 score
// stripe lives in LDS (256KB of the 320KB/WGP on CDNA5) -> exact softmax,
// scores never written to HBM.
__global__ void attn_wmma(const float* __restrict__ qkv, float* __restrict__ o)
{
  extern __shared__ float S[];                // 16 * 4096 floats
  __shared__ float rowM[16], rowSum[16];
  const int h     = blockIdx.y;
  const int mBase = blockIdx.x * 16;
  const int lane  = threadIdx.x & 31;
  const int wave  = threadIdx.x >> 5;
  const int ld    = 3 * GD;

  const float* Qrow = qkv + (size_t)(mBase + (lane & 15)) * ld + h * DH;

  // Phase A: S[16,4096] = (Q @ K^T) * 1/sqrt(dh)
  for (int kt = wave; kt < GN / 16; kt += 4) {
    const int keyBase = kt * 16;
    const float* Krow = qkv + (size_t)(keyBase + (lane & 15)) * ld + GD + h * DH;
    v8f acc = {};
#pragma unroll
    for (int kd = 0; kd < DH; kd += 32) {
      v16bf a = frag_rowK(Qrow, kd, lane);
      v16bf b = frag_rowK(Krow, kd, lane);
      acc = __builtin_amdgcn_wmma_f32_16x16x32_bf16(false, a, false, b,
                                                    (short)0, acc, false, false);
    }
    const int rofs = (lane < 16) ? 0 : 8;
#pragma unroll
    for (int r = 0; r < 8; ++r)
      S[(size_t)(rofs + r) * GN + keyBase + (lane & 15)] = acc[r] * 0.125f;
  }
  __syncthreads();

  // Phase B: exact softmax per row
  if (threadIdx.x < 16) {
    const int r = threadIdx.x;
    float mx = -3.4e38f;
    for (int j = 0; j < GN; ++j) mx = fmaxf(mx, S[(size_t)r * GN + j]);
    float sm = 0.f;
    for (int j = 0; j < GN; ++j) sm += __expf(S[(size_t)r * GN + j] - mx);
    rowM[r] = mx;
    rowSum[r] = 1.f / (sm + 1e-20f);
  }
  __syncthreads();
  for (int j = threadIdx.x; j < 16 * GN; j += 128) {
    int r = j >> 12;                          // GN == 2^12
    S[j] = __expf(S[j] - rowM[r]) * rowSum[r];
  }
  __syncthreads();

  // Phase C: O[16,64] = P @ V ; wave w handles output cols [16w, 16w+16)
  const int dBase = wave * 16;
  const float* Srow = S + (size_t)(lane & 15) * GN;
  const float* Vcol = qkv + 2 * GD + h * DH + dBase + (lane & 15);
  v8f acc = {};
  for (int kc = 0; kc < GN; kc += 32) {
    v16bf a = frag_rowK(Srow, kc, lane);
    v16bf b = frag_colK(Vcol, ld, kc, lane);
    acc = __builtin_amdgcn_wmma_f32_16x16x32_bf16(false, a, false, b,
                                                  (short)0, acc, false, false);
  }
  const int rofs = (lane < 16) ? 0 : 8;
#pragma unroll
  for (int r = 0; r < 8; ++r)
    o[(size_t)(mBase + rofs + r) * GD + h * DH + dBase + (lane & 15)] = acc[r];
}

// ------------------------------- LayerNorm ---------------------------------
__global__ void layernorm_k(const float* __restrict__ in, const float* __restrict__ g,
                            const float* __restrict__ b, float* __restrict__ out, int D)
{
  __shared__ float red[256];
  const int row = blockIdx.x, t = threadIdx.x;
  float v = in[(size_t)row * D + t];
  red[t] = v; __syncthreads();
  for (int s = 128; s > 0; s >>= 1) { if (t < s) red[t] += red[t + s]; __syncthreads(); }
  float mu = red[0] / D; __syncthreads();
  float d = v - mu; red[t] = d * d; __syncthreads();
  for (int s = 128; s > 0; s >>= 1) { if (t < s) red[t] += red[t + s]; __syncthreads(); }
  float var = red[0] / D;
  out[(size_t)row * D + t] = d * rsqrtf(var + 1e-5f) * g[t] + b[t];
}

// ------------------------------- GAT pieces --------------------------------
__device__ __forceinline__ unsigned fenc(float f) {
  unsigned u = __float_as_uint(f);
  return (u >> 31) ? ~u : (u | 0x80000000u);
}
__device__ __forceinline__ float fdec(unsigned u) {
  return (u >> 31) ? __uint_as_float(u ^ 0x80000000u) : __uint_as_float(~u);
}

__global__ void gat_coeff_k(const float* __restrict__ hbuf, const float* __restrict__ asrc,
                            const float* __restrict__ adst, float* __restrict__ as,
                            float* __restrict__ ad)
{
  int i = blockIdx.x * blockDim.x + threadIdx.x;
  if (i >= GN * GH) return;
  int n = i >> 2, hh = i & 3;
  const float* hp = hbuf + (size_t)n * (GH * GC) + hh * GC;
  const float* s  = asrc + hh * GC;
  const float* d  = adst + hh * GC;
  float s0 = 0.f, d0 = 0.f;
  for (int c = 0; c < GC; ++c) { float v = hp[c]; s0 += v * s[c]; d0 += v * d[c]; }
  as[i] = s0; ad[i] = d0;
}

__global__ void edge_logits_k(const int* __restrict__ ei, const float* __restrict__ as,
                              const float* __restrict__ ad, float* __restrict__ ew,
                              unsigned* __restrict__ menc)
{
  int e = blockIdx.x * blockDim.x + threadIdx.x;
  const int Etot = GE + GN;
  if (e >= Etot) return;
  int s = (e < GE) ? ei[e] : (e - GE);
  int d = (e < GE) ? ei[GE + e] : (e - GE);
#pragma unroll
  for (int hh = 0; hh < GH; ++hh) {
    float v = as[s * GH + hh] + ad[d * GH + hh];
    v = (v > 0.f) ? v : 0.2f * v;                 // leaky_relu 0.2
    ew[(size_t)e * GH + hh] = v;
    atomicMax(&menc[d * GH + hh], fenc(v));       // segment max
  }
}

__global__ void edge_exp_k(const int* __restrict__ ei, float* __restrict__ ew,
                           const unsigned* __restrict__ menc, float* __restrict__ denom)
{
  int e = blockIdx.x * blockDim.x + threadIdx.x;
  const int Etot = GE + GN;
  if (e >= Etot) return;
  int d = (e < GE) ? ei[GE + e] : (e - GE);
#pragma unroll
  for (int hh = 0; hh < GH; ++hh) {
    float w = __expf(ew[(size_t)e * GH + hh] - fdec(menc[d * GH + hh]));
    ew[(size_t)e * GH + hh] = w;
    atomicAdd(&denom[d * GH + hh], w);
  }
}

__global__ void edge_agg_k(const int* __restrict__ ei, const float* __restrict__ ew,
                           const float* __restrict__ denom, const float* __restrict__ hbuf,
                           float* __restrict__ agg)
{
  int e = blockIdx.x;
  const int Etot = GE + GN;
  if (e >= Etot) return;
  int s = (e < GE) ? ei[e] : (e - GE);
  int d = (e < GE) ? ei[GE + e] : (e - GE);
  for (int j = threadIdx.x; j < GH * GC; j += 256) {
    int hh = j >> 8;                              // GC == 256
    float alpha = ew[(size_t)e * GH + hh] / (denom[d * GH + hh] + 1e-16f);
    atomicAdd(&agg[(size_t)d * (GH * GC) + j], hbuf[(size_t)s * (GH * GC) + j] * alpha);
  }
}

__global__ void head_mean_k(const float* __restrict__ agg, const float* __restrict__ gbias,
                            float* __restrict__ xout)
{
  int i = blockIdx.x * blockDim.x + threadIdx.x;
  if (i >= GN * GC) return;
  int n = i >> 8, c = i & 255;
  const float* a = agg + (size_t)n * (GH * GC);
  xout[i] = 0.25f * (a[c] + a[GC + c] + a[2 * GC + c] + a[3 * GC + c]) + gbias[c];
}

// -------------------------------- utils ------------------------------------
__global__ void fill_f(float* p, long n, float v) {
  long i = (long)blockIdx.x * 256 + threadIdx.x;
  if (i < n) p[i] = v;
}
__global__ void fill_u(unsigned* p, long n, unsigned v) {
  long i = (long)blockIdx.x * 256 + threadIdx.x;
  if (i < n) p[i] = v;
}

// ------------------------------ launcher -----------------------------------
extern "C" void kernel_launch(void* const* d_in, const int* in_sizes, int n_in,
                              void* d_out, int out_size, void* d_ws, size_t ws_size,
                              hipStream_t stream) {
  const float* x_in   = (const float*)d_in[0];
  const int*   ei     = (const int*)d_in[1];
  const float* gat_W  = (const float*)d_in[2];
  const float* gat_as = (const float*)d_in[3];
  const float* gat_ad = (const float*)d_in[4];
  const float* gat_b  = (const float*)d_in[5];
  const float* qkv_w  = (const float*)d_in[6];
  const float* qkv_b  = (const float*)d_in[7];
  const float* out_w  = (const float*)d_in[8];
  const float* out_b  = (const float*)d_in[9];
  const float* ff1_w  = (const float*)d_in[10];
  const float* ff1_b  = (const float*)d_in[11];
  const float* ff2_w  = (const float*)d_in[12];
  const float* ff2_b  = (const float*)d_in[13];
  const float* ln1_w  = (const float*)d_in[14];
  const float* ln1_b  = (const float*)d_in[15];
  const float* ln2_w  = (const float*)d_in[16];
  const float* ln2_b  = (const float*)d_in[17];

  // ----- workspace carve -----
  char* base = (char*)d_ws;
  size_t off = 0;
  auto carve = [&](size_t nfloats) -> float* {
    float* p = (float*)(base + off);
    off += nfloats * sizeof(float);
    return p;
  };
  float*    xa    = carve((size_t)GN * GC);
  float*    xb    = carve((size_t)GN * GC);
  float*    hbuf  = carve((size_t)GN * GH * GC);      // also FF intermediate (with agg)
  float*    agg   = carve((size_t)GN * GH * GC);
  float*    as    = carve((size_t)GN * GH);
  float*    ad    = carve((size_t)GN * GH);
  unsigned* menc  = (unsigned*)carve((size_t)GN * GH);
  float*    denom = carve((size_t)GN * GH);
  float*    ew    = carve((size_t)(GE + GN) * GH);
  float*    qkvb  = carve((size_t)GN * 3 * GD);
  float*    obuf  = carve((size_t)GN * GD);
  float*    tmp   = carve((size_t)GN * GD);
  float*    f1    = hbuf;   // N*FF floats == hbuf+agg span (8M floats), GAT phase done
  if (off > ws_size) return;

  const int  Etot = GE + GN;
  const long aggN = (long)GN * GH * GC;

  (void)hipMemcpyAsync(xa, x_in, (size_t)GN * GC * sizeof(float),
                       hipMemcpyDeviceToDevice, stream);

  // ================= stacked GAT layers =================
  for (int l = 0; l < GL; ++l) {
    // h = x @ gat_W[l]   [4096,256] x [256,1024]  (WMMA bf16, TDM-staged A)
    gemm_bf16_wmma<<<dim3(GN / 16, (GH * GC) / 256), 128, 16 * GC * sizeof(float), stream>>>(
        xa, gat_W + (size_t)l * GC * GH * GC, nullptr, nullptr, hbuf,
        GN, GH * GC, GC, /*btrans=*/0, /*relu=*/0);

    gat_coeff_k<<<(GN * GH + 255) / 256, 256, 0, stream>>>(
        hbuf, gat_as + (size_t)l * GH * GC, gat_ad + (size_t)l * GH * GC, as, ad);

    fill_u<<<(GN * GH + 255) / 256, 256, 0, stream>>>(menc, GN * GH, 0x007FFFFFu); // enc(-inf)
    fill_f<<<(GN * GH + 255) / 256, 256, 0, stream>>>(denom, GN * GH, 0.f);
    fill_f<<<(int)((aggN + 255) / 256), 256, 0, stream>>>(agg, aggN, 0.f);

    edge_logits_k<<<(Etot + 255) / 256, 256, 0, stream>>>(ei, as, ad, ew, menc);
    edge_exp_k<<<(Etot + 255) / 256, 256, 0, stream>>>(ei, ew, menc, denom);
    edge_agg_k<<<Etot, 256, 0, stream>>>(ei, ew, denom, hbuf, agg);

    head_mean_k<<<(GN * GC + 255) / 256, 256, 0, stream>>>(agg, gat_b + (size_t)l * GC, xb);
    float* t = xa; xa = xb; xb = t;
  }

  // ================= transformer encoder layers =================
  for (int l = 0; l < GL; ++l) {
    // qkv = x @ qkv_w^T + b   [4096,256] x [256,768]
    gemm_bf16_wmma<<<dim3(GN / 16, (3 * GD) / 256), 128, 16 * GD * sizeof(float), stream>>>(
        xa, qkv_w + (size_t)l * 3 * GD * GD, qkv_b + (size_t)l * 3 * GD, nullptr, qkvb,
        GN, 3 * GD, GD, 1, 0);

    // fused multi-head attention, 256KB LDS score stripe per workgroup
    attn_wmma<<<dim3(GN / 16, GH), 128, 16 * GN * sizeof(float), stream>>>(qkvb, obuf);

    // o @ out_w^T + out_b + x (residual)
    gemm_bf16_wmma<<<dim3(GN / 16, GD / 256), 128, 16 * GD * sizeof(float), stream>>>(
        obuf, out_w + (size_t)l * GD * GD, out_b + (size_t)l * GD, xa, tmp,
        GN, GD, GD, 1, 0);
    layernorm_k<<<GN, 256, 0, stream>>>(tmp, ln1_w + (size_t)l * GD, ln1_b + (size_t)l * GD, xb, GD);

    // f1 = relu(x @ ff1_w^T + b)   [4096,256] x [256,2048]
    gemm_bf16_wmma<<<dim3(GN / 16, GFF / 256), 128, 16 * GD * sizeof(float), stream>>>(
        xb, ff1_w + (size_t)l * GFF * GD, ff1_b + (size_t)l * GFF, nullptr, f1,
        GN, GFF, GD, 1, 1);
    // f1 @ ff2_w^T + b + x (residual)  [4096,2048] x [2048,256]   (128KB A panel)
    gemm_bf16_wmma<<<dim3(GN / 16, GD / 256), 128, 16 * GFF * sizeof(float), stream>>>(
        f1, ff2_w + (size_t)l * GD * GFF, ff2_b + (size_t)l * GD, xb, tmp,
        GN, GD, GFF, 1, 0);

    float* dst = (l == GL - 1) ? (float*)d_out : xa;
    layernorm_k<<<GN, 256, 0, stream>>>(tmp, ln2_w + (size_t)l * GD, ln2_b + (size_t)l * GD, dst, GD);
  }
}
